// Attention_55430847922507
// MI455X (gfx1250) — matrix operand order repaired and checked by reading, not compiled, not run on hardware
//
#include <hip/hip_runtime.h>
#include <hip/hip_bf16.h>

// ---------------------------------------------------------------------------
// Attention (Bahdanau) on gfx1250:
//   energy = tanh(features@Wf + (hidden@Wh + b)[:,None,:])
//   scores = energy @ v ; w = softmax(scores, axis=L) ; context = w^T features
// features: (64, 2048, 512) f32; W_att: (1024,512); out = [context | weights]
// ---------------------------------------------------------------------------

typedef __attribute__((ext_vector_type(16))) __bf16 v16bf;
typedef __attribute__((ext_vector_type(8)))  __bf16 v8bf;
typedef __attribute__((ext_vector_type(8)))  float  v8f;

#define B_DIM 64
#define L_DIM 2048
#define F_DIM 512
#define H_DIM 512
#define BL    (B_DIM * L_DIM)          // 131072 rows
#define TILE_M 128                     // rows per workgroup (8 waves x 16)
#define LDS_ROW 520                    // 512 + 8 bf16 pad (16B) vs bank conflicts

// Native V_TANH_F32 if the toolchain exposes it (gfx1250 TRANS op),
// else fall back to libm tanhf.
#if defined(__has_builtin)
#if __has_builtin(__builtin_amdgcn_tanhf)
#define FAST_TANH(x) __builtin_amdgcn_tanhf(x)
#endif
#endif
#ifndef FAST_TANH
#define FAST_TANH(x) tanhf(x)
#endif

// ---------------- prep: Wf (f32, [k][n]) -> Wt (bf16, [n][k]) ---------------
__global__ void wt_kernel(const float* __restrict__ W_att, __bf16* __restrict__ Wt) {
    int idx = blockIdx.x * 256 + threadIdx.x;      // 0 .. 512*512-1
    int n = idx >> 9;
    int k = idx & 511;
    Wt[(size_t)n * F_DIM + k] = (__bf16)W_att[(size_t)k * H_DIM + n];
}

// ---------------- prep: hb[b][h] = hidden@Wh + b_att ------------------------
__global__ void hb_kernel(const float* __restrict__ hidden,
                          const float* __restrict__ W_att,
                          const float* __restrict__ b_att,
                          float* __restrict__ hb) {
    int b = blockIdx.x;
    int tid = threadIdx.x;
    #pragma unroll
    for (int j = 0; j < 2; ++j) {
        int h = tid + j * 256;
        float acc = b_att[h];
        for (int f = 0; f < F_DIM; ++f)
            acc += hidden[b * F_DIM + f] * W_att[(size_t)(F_DIM + f) * H_DIM + h];
        hb[b * H_DIM + h] = acc;
    }
}

// ---------------- main: WMMA bf16 GEMM + tanh + dot(v) -> scores ------------
__global__ void scores_kernel(const float* __restrict__ feats,
                              const __bf16* __restrict__ Wt,
                              const float* __restrict__ hb,
                              const float* __restrict__ vvec,
                              float* __restrict__ scores) {
    extern __shared__ __bf16 lds_a[];              // [TILE_M][LDS_ROW] bf16

    const int tid  = threadIdx.x;
    const int wave = tid >> 5;
    const int lane = tid & 31;
    const int m0_blk = blockIdx.x * TILE_M;        // first global row of tile
    const int b = m0_blk >> 11;                    // row / 2048 (same for tile)

    // ---- stage A tile: 128 rows x 512 f32 -> bf16 LDS (coalesced b128) ----
    const float4* gsrc = (const float4*)(feats + (size_t)m0_blk * F_DIM);
    #pragma unroll 4
    for (int i = 0; i < 64; ++i) {
        int idx  = tid + i * 256;                  // float4 index in tile
        int row  = idx >> 7;                       // 128 float4 per row
        int col4 = idx & 127;
        float4 val = gsrc[idx];
        __bf16* dst = lds_a + row * LDS_ROW + col4 * 4;
        dst[0] = (__bf16)val.x; dst[1] = (__bf16)val.y;
        dst[2] = (__bf16)val.z; dst[3] = (__bf16)val.w;
    }
    __syncthreads();

    // ---- per-wave 16-row subtile -------------------------------------------
    const int mrow  = wave * 16 + (lane & 15);     // A row this lane feeds
    const int hi    = lane >> 4;                   // lane half select
    const int hi8   = hi << 4;                     // A K-offset: 0 / 16
    const int nlane = lane & 15;                   // N index within tile
    const __bf16* aptr = lds_a + mrow * LDS_ROW + hi8;

    float score[8];
    #pragma unroll
    for (int r = 0; r < 8; ++r) score[r] = 0.0f;

    // 8 groups of 4 N-tiles: 4 C accumulators (32 VGPR), 1 A frag shared by
    // 4 WMMAs per k-step. The asm memory clobber pins the LDS A reads inside
    // this loop (otherwise LICM hoists 16 fragments -> 128 VGPRs -> spills).
    for (int ng = 0; ng < 8; ++ng) {
        asm volatile("" ::: "memory");
        const __bf16* bp = Wt + (size_t)(ng * 64 + nlane) * F_DIM + (hi << 4);
        if (ng < 7)                                 // pull next N-group of Wt
            __builtin_prefetch((const void*)(bp + (size_t)64 * F_DIM), 0, 1);

        v8f c[4];
        #pragma unroll
        for (int j = 0; j < 4; ++j) c[j] = (v8f){};

        #pragma unroll 4
        for (int k0 = 0; k0 < F_DIM; k0 += 32) {
            // A fragment (16x32 bf16): two 16B LDS reads, contiguous k = 16*hi + s
            v8bf a0 = *(const v8bf*)(aptr + k0);
            v8bf a1 = *(const v8bf*)(aptr + k0 + 8);
            v16bf a;
            #pragma unroll
            for (int i = 0; i < 8; ++i) { a[i] = a0[i]; a[8 + i] = a1[i]; }
            // 4 independent WMMAs reuse the A fragment
            #pragma unroll
            for (int j = 0; j < 4; ++j) {
                v16bf bm = *(const v16bf*)(bp + (size_t)j * 16 * F_DIM + k0);
                c[j] = __builtin_amdgcn_wmma_f32_16x16x32_bf16(
                           false, a, false, bm, (short)0, c[j], false, false);
            }
        }

        // energy = tanh(c + hb), partial score += energy * v[n]
        #pragma unroll
        for (int j = 0; j < 4; ++j) {
            int n = ng * 64 + j * 16 + nlane;
            float hbv = hb[b * H_DIM + n];
            float vv  = vvec[n];
            #pragma unroll
            for (int r = 0; r < 8; ++r)
                score[r] += FAST_TANH(c[j][r] + hbv) * vv;
        }
    }

    // reduce over the 16 N-lanes of each half-wave
    #pragma unroll
    for (int r = 0; r < 8; ++r) {
        float s = score[r];
        s += __shfl_xor(s, 8, 32);
        s += __shfl_xor(s, 4, 32);
        s += __shfl_xor(s, 2, 32);
        s += __shfl_xor(s, 1, 32);
        score[r] = s;
    }
    if ((lane & 15) == 0) {
        int mbase = m0_blk + wave * 16 + (hi ? 8 : 0);   // C layout: M = r + 8*hi
        #pragma unroll
        for (int r = 0; r < 8; ++r) scores[mbase + r] = score[r];
    }
}

// ---------------- softmax over L per batch ----------------------------------
__global__ void softmax_kernel(const float* __restrict__ scores,
                               float* __restrict__ out_w) {
    __shared__ float red[256];
    int b = blockIdx.x, tid = threadIdx.x;
    float s[8];
    float mx = -3.402823466e38f;
    #pragma unroll
    for (int i = 0; i < 8; ++i) {
        s[i] = scores[b * L_DIM + tid + i * 256];
        mx = fmaxf(mx, s[i]);
    }
    red[tid] = mx; __syncthreads();
    for (int st = 128; st > 0; st >>= 1) {
        if (tid < st) red[tid] = fmaxf(red[tid], red[tid + st]);
        __syncthreads();
    }
    mx = red[0]; __syncthreads();

    float sum = 0.0f;
    #pragma unroll
    for (int i = 0; i < 8; ++i) { s[i] = __expf(s[i] - mx); sum += s[i]; }
    red[tid] = sum; __syncthreads();
    for (int st = 128; st > 0; st >>= 1) {
        if (tid < st) red[tid] += red[tid + st];
        __syncthreads();
    }
    float inv = 1.0f / red[0];
    #pragma unroll
    for (int i = 0; i < 8; ++i)
        out_w[b * L_DIM + tid + i * 256] = s[i] * inv;
}

// ---------------- context = w^T features (per batch) ------------------------
__global__ void zero_ctx_kernel(float* __restrict__ ctx) {
    ctx[blockIdx.x * 256 + threadIdx.x] = 0.0f;
}

__global__ void context_kernel(const float* __restrict__ feats,
                               const float* __restrict__ w,
                               float* __restrict__ ctx) {
    int b  = blockIdx.x >> 3;
    int lc = blockIdx.x & 7;
    int tid = threadIdx.x;
    int l0 = lc * 256;
    const float* fb = feats + ((size_t)b * L_DIM + l0) * F_DIM;
    const float* wb = w + b * L_DIM + l0;
    float acc0 = 0.0f, acc1 = 0.0f;
    for (int l = 0; l < 256; ++l) {
        float wv = wb[l];
        acc0 += wv * fb[(size_t)l * F_DIM + tid];
        acc1 += wv * fb[(size_t)l * F_DIM + tid + 256];
    }
    atomicAdd(&ctx[b * F_DIM + tid],       acc0);
    atomicAdd(&ctx[b * F_DIM + tid + 256], acc1);
}

// ---------------------------------------------------------------------------
extern "C" void kernel_launch(void* const* d_in, const int* in_sizes, int n_in,
                              void* d_out, int out_size, void* d_ws, size_t ws_size,
                              hipStream_t stream) {
    const float* feats  = (const float*)d_in[0];   // (64,2048,512)
    const float* hidden = (const float*)d_in[1];   // (64,512)
    const float* W_att  = (const float*)d_in[2];   // (1024,512)
    const float* b_att  = (const float*)d_in[3];   // (512,)
    const float* vvec   = (const float*)d_in[4];   // (512,)

    float* out     = (float*)d_out;                // [context(32768) | weights(131072)]
    float* scores  = (float*)d_ws;                 // 131072 f32
    float* hb      = scores + BL;                  // 32768 f32
    __bf16* Wt     = (__bf16*)(hb + B_DIM * H_DIM);// 262144 bf16

    wt_kernel<<<(F_DIM * H_DIM) / 256, 256, 0, stream>>>(W_att, Wt);
    hb_kernel<<<B_DIM, 256, 0, stream>>>(hidden, W_att, b_att, hb);
    zero_ctx_kernel<<<(B_DIM * F_DIM) / 256, 256, 0, stream>>>(out);

    size_t lds_bytes = (size_t)TILE_M * LDS_ROW * sizeof(__bf16);  // ~130 KB
    scores_kernel<<<BL / TILE_M, 256, lds_bytes, stream>>>(feats, Wt, hb, vvec, scores);

    softmax_kernel<<<B_DIM, 256, 0, stream>>>(scores, out + B_DIM * F_DIM);
    context_kernel<<<B_DIM * 8, 256, 0, stream>>>(feats, out + B_DIM * F_DIM, out);
}
